// ImplicitDecoder_17471926960966
// MI455X (gfx1250) — matrix operand order repaired and checked by reading, not compiled, not run on hardware
//
#include <hip/hip_runtime.h>
#include <hip/hip_bf16.h>
#include <stdint.h>
#include <stddef.h>

typedef __attribute__((ext_vector_type(16))) __bf16 v16bf;
typedef __attribute__((ext_vector_type(8)))  __bf16 v8bf;
typedef __attribute__((ext_vector_type(8)))  float  v8f;

#define HID 256
#define FNUM 800
#define TILE_PTS 32
#define NWAVES 8
#define FSTRIDE 808   // 800 + 8 pad (bf16 elems)
#define N0STRIDE 520  // 512 + 8
#define N1STRIDE 264  // 256 + 8

#define SMEM_A_BYTES (TILE_PTS * FSTRIDE * 2)     // 51712 (feats -> later net1)
#define SMEM_B_BYTES (TILE_PTS * N0STRIDE * 2)    // 33280 (net0 -> later net2)
#define SMEM_TOTAL   (SMEM_A_BYTES + SMEM_B_BYTES)

// ---------------------------------------------------------------------------
// 1) Volume transpose: [C, V] -> [V, C]  (V = S^3), tiled through LDS
// ---------------------------------------------------------------------------
__global__ __launch_bounds__(256)
void transpose_cv_kernel(const float* __restrict__ in, float* __restrict__ out,
                         int C, int V) {
  __shared__ float tile[32][33];
  const int vbase = blockIdx.x * 32;
  const int cbase = blockIdx.y * 32;
  const int tx = threadIdx.x;   // 0..31
  const int ty = threadIdx.y;   // 0..7
#pragma unroll
  for (int r = 0; r < 32; r += 8) {
    int c = cbase + ty + r;
    int v = vbase + tx;
    float val = 0.0f;
    if (c < C && v < V) val = in[(size_t)c * V + v];
    tile[ty + r][tx] = val;
  }
  __syncthreads();
#pragma unroll
  for (int r = 0; r < 32; r += 8) {
    int v = vbase + ty + r;
    int c = cbase + tx;
    if (c < C && v < V) out[(size_t)v * C + c] = tile[tx][ty + r];
  }
}

// ---------------------------------------------------------------------------
// 2) Weight repack: fp32 [Cout, Cin] row-major -> bf16 WMMA B-fragments.
//    Tile (nt, kt) holds 512 bf16: lane L (0..31) owns elems [L*16 .. L*16+15].
//    B layout (16-bit, 32x16): N = L%16; element e -> K = (e/8)*16 + (L/16)*8 + e%8.
//    tile index = nt*KT + kt  (k-tiles contiguous for streaming in hot loop)
// ---------------------------------------------------------------------------
__global__ __launch_bounds__(256)
void repack_w_kernel(const float* __restrict__ w, __bf16* __restrict__ out,
                     int CoutReal, int Cin, int KT, int total) {
  int i = blockIdx.x * 256 + threadIdx.x;
  if (i >= total) return;
  int tile = i >> 9;          // /512
  int r    = i & 511;
  int lane = r >> 4;
  int e    = r & 15;
  int kt   = tile % KT;
  int nt   = tile / KT;
  int n    = nt * 16 + (lane & 15);
  int k    = kt * 32 + (e >> 3) * 16 + (lane >> 4) * 8 + (e & 7);
  float val = (n < CoutReal) ? w[(size_t)n * Cin + k] : 0.0f;
  out[i] = (__bf16)val;
}

// ---------------------------------------------------------------------------
// One MLP layer on a 32-point tile. Each wave owns an N tile (16 outputs) and
// computes BOTH 16-row M tiles per K step, so every B fragment is reused for
// two wmmas (2x arithmetic intensity on the weight stream).
// The asm compiler barrier at the top of each nt iteration stops LICM from
// hoisting the (nt-invariant) A-fragment LDS loads out of the job loop:
// re-reading A from LDS per job is cheap, while hoisting either bloats VGPRs
// to ~450 (kills occupancy for the latency-bound gather phase) or spills to
// scratch (global-backed — worse than the traffic it saves).
// A-fragment (16-bit 16x32): M = lane%16, K = (e<8 ? 0 : 16) + (lane/16)*8 + e%8
//   -> two contiguous 8-bf16 (16B) LDS reads per k-tile.
// D layout (f32 16x16): N = lane%16, M = vgpr + 8*(lane/16).
// ---------------------------------------------------------------------------
__device__ __forceinline__ void mlp_layer_wmma(
    const __bf16* __restrict__ inbuf, int istride,
    __bf16* __restrict__ outbuf, int ostride,
    const __bf16* __restrict__ wrep, const float* __restrict__ bias,
    int Cin, int Cout, int lane, int wave) {
  const int KT   = Cin >> 5;
  const int NT   = Cout >> 4;
  const int half = lane >> 4;
  const int col  = lane & 15;
  for (int nt = wave; nt < NT; nt += NWAVES) {
    asm volatile("" ::: "memory");  // compiler barrier: keep A loads in-loop
    const int n0 = nt << 4;
    const float bv = bias[n0 + col];
    v8f acc0 = {bv, bv, bv, bv, bv, bv, bv, bv};
    v8f acc1 = acc0;
    const __bf16* arow0 = inbuf + col * istride + half * 8;          // M tile 0
    const __bf16* arow1 = inbuf + (16 + col) * istride + half * 8;   // M tile 1
    const __bf16* btile = wrep + (size_t)(nt * KT) * 512 + lane * 16;
#pragma unroll 4
    for (int kt = 0; kt < KT; ++kt) {
      v16bf bm = *(const v16bf*)(btile + (size_t)kt * 512);
      union { v16bf v; v8bf h[2]; } a0, a1;
      a0.h[0] = *(const v8bf*)(arow0 + kt * 32);
      a0.h[1] = *(const v8bf*)(arow0 + kt * 32 + 16);
      a1.h[0] = *(const v8bf*)(arow1 + kt * 32);
      a1.h[1] = *(const v8bf*)(arow1 + kt * 32 + 16);
      acc0 = __builtin_amdgcn_wmma_f32_16x16x32_bf16(
          false, a0.v, false, bm, (short)0, acc0, false, false);
      acc1 = __builtin_amdgcn_wmma_f32_16x16x32_bf16(
          false, a1.v, false, bm, (short)0, acc1, false, false);
    }
    __bf16* orow = outbuf + n0 + col;
#pragma unroll
    for (int v = 0; v < 8; ++v) {
      float v0 = acc0[v] > 0.0f ? acc0[v] : 0.0f;  // ReLU
      float v1 = acc1[v] > 0.0f ? acc1[v] : 0.0f;
      orow[(half * 8 + v) * ostride]        = (__bf16)v0;
      orow[(16 + half * 8 + v) * ostride]   = (__bf16)v1;
    }
  }
}

// ---------------------------------------------------------------------------
// 3) Fused: sample 800 feats -> LDS (bf16) -> 3 WMMA layers -> VALU head
// ---------------------------------------------------------------------------
__global__ __launch_bounds__(256)
void decoder_fused_kernel(
    const float* __restrict__ vt0, const float* __restrict__ vt1,
    const float* __restrict__ vt2, const float* __restrict__ vt3,
    const float* __restrict__ vt4, const float* __restrict__ pts,
    const __bf16* __restrict__ w0r, const __bf16* __restrict__ w1r,
    const __bf16* __restrict__ w2r,
    const float* __restrict__ b0, const float* __restrict__ b1,
    const float* __restrict__ b2,
    const float* __restrict__ w3, const float* __restrict__ b3,
    float* __restrict__ out, int N) {
  __shared__ __align__(16) char smem[SMEM_TOTAL];
  __bf16* regA = (__bf16*)smem;                    // feats, later net1
  __bf16* regB = (__bf16*)(smem + SMEM_A_BYTES);   // net0, later net2

  const int tid   = threadIdx.x;
  const int nbase = blockIdx.x * TILE_PTS;
  const int lane  = tid & 31;
  const int wave  = tid >> 5;

  // ---- sampling: 8 threads per point, each owns 4 consecutive channels ----
  {
    const int p   = tid >> 3;
    const int sub = tid & 7;
    const int n   = nbase + p;
    float px = 0.f, py = 0.f, pz = 0.f;
    if (n < N) { px = pts[n * 3 + 0]; py = pts[n * 3 + 1]; pz = pts[n * 3 + 2]; }

    const int   Ss[5]   = {128, 64, 32, 16, 8};
    const int   Cs[5]   = {32, 64, 128, 256, 320};
    const int   Offs[5] = {0, 32, 96, 224, 480};
    const float* vts[5] = {vt0, vt1, vt2, vt3, vt4};

    for (int l = 0; l < 5; ++l) {
      const int S = Ss[l], C = Cs[l], coff = Offs[l];
      const float* __restrict__ vt = vts[l];
      // align_corners=False unnormalize
      float xs = (px + 1.f) * 0.5f * (float)S - 0.5f;
      float ys = (py + 1.f) * 0.5f * (float)S - 0.5f;
      float zs = (pz + 1.f) * 0.5f * (float)S - 0.5f;
      float xf = floorf(xs), yf = floorf(ys), zf = floorf(zs);
      int ix = (int)xf, iy = (int)yf, iz = (int)zf;
      float fx = xs - xf, fy = ys - yf, fz = zs - zf;
      int   offc[8];
      float wc[8];
#pragma unroll
      for (int j = 0; j < 8; ++j) {
        int dz = (j >> 2) & 1, dy = (j >> 1) & 1, dx = j & 1;
        int zz = iz + dz, yy = iy + dy, xx = ix + dx;
        bool ok = (zz >= 0) && (zz < S) && (yy >= 0) && (yy < S) &&
                  (xx >= 0) && (xx < S);
        float w = (dz ? fz : 1.f - fz) * (dy ? fy : 1.f - fy) *
                  (dx ? fx : 1.f - fx);
        offc[j] = ok ? ((zz * S + yy) * S + xx) * C : 0;
        wc[j]   = ok ? w : 0.f;
      }
      // 4 channels per float4 gather per corner; C % 32 == 0 for all levels
      for (int c = sub * 4; c < C; c += 32) {
        float ax = 0.f, ay = 0.f, az = 0.f, aw = 0.f;
#pragma unroll
        for (int j = 0; j < 8; ++j) {
          const float4 v = *(const float4*)(vt + offc[j] + c);
          ax += wc[j] * v.x; ay += wc[j] * v.y;
          az += wc[j] * v.z; aw += wc[j] * v.w;
        }
        __bf16* dst = regA + p * FSTRIDE + coff + c;
        dst[0] = (__bf16)ax; dst[1] = (__bf16)ay;
        dst[2] = (__bf16)az; dst[3] = (__bf16)aw;
      }
    }
  }
  __syncthreads();

  // L0: feats(regA, 800) -> net0(regB, 512)
  mlp_layer_wmma(regA, FSTRIDE, regB, N0STRIDE, w0r, b0, FNUM, 2 * HID, lane, wave);
  __syncthreads();
  // L1: net0(regB, 512) -> net1(regA, 256)
  mlp_layer_wmma(regB, N0STRIDE, regA, N1STRIDE, w1r, b1, 2 * HID, HID, lane, wave);
  __syncthreads();
  // L2: net1(regA, 256) -> net2(regB, 256)
  mlp_layer_wmma(regA, N1STRIDE, regB, N1STRIDE, w2r, b2, HID, HID, lane, wave);
  __syncthreads();

  // L3 head: 3x256, VALU dot products (no relu)
  if (tid < TILE_PTS * 3) {
    const int p = tid / 3, o = tid % 3;
    const int n = nbase + p;
    if (n < N) {
      const __bf16* row = regB + p * N1STRIDE;
      const float* wr = w3 + o * HID;
      float acc = b3[o];
#pragma unroll 8
      for (int c = 0; c < HID; ++c) acc += (float)row[c] * wr[c];
      out[(size_t)o * N + n] = acc;
    }
  }
}

// ---------------------------------------------------------------------------
extern "C" void kernel_launch(void* const* d_in, const int* in_sizes, int n_in,
                              void* d_out, int out_size, void* d_ws, size_t ws_size,
                              hipStream_t stream) {
  const float* skip[5];
  for (int i = 0; i < 5; ++i) skip[i] = (const float*)d_in[i];
  const float* pts = (const float*)d_in[5];
  const float* w0 = (const float*)d_in[6];
  const float* b0 = (const float*)d_in[7];
  const float* w1 = (const float*)d_in[8];
  const float* b1 = (const float*)d_in[9];
  const float* w2 = (const float*)d_in[10];
  const float* b2 = (const float*)d_in[11];
  const float* w3 = (const float*)d_in[12];
  const float* b3 = (const float*)d_in[13];
  float* out = (float*)d_out;
  const int N = in_sizes[5] / 3;

  const int Cs[5] = {32, 64, 128, 256, 320};
  const int Ss[5] = {128, 64, 32, 16, 8};

  // workspace layout
  char* ws = (char*)d_ws;
  size_t off = 0;
  float* vt[5];
  for (int l = 0; l < 5; ++l) {
    vt[l] = (float*)(ws + off);
    off += (size_t)Cs[l] * Ss[l] * Ss[l] * Ss[l] * sizeof(float);
  }
  __bf16* w0r = (__bf16*)(ws + off); off += (size_t)(2 * HID) * FNUM * 2;      // 819200
  __bf16* w1r = (__bf16*)(ws + off); off += (size_t)HID * (2 * HID) * 2;       // 262144
  __bf16* w2r = (__bf16*)(ws + off); off += (size_t)HID * HID * 2;             // 131072

  // 1) transpose volumes [C,V] -> [V,C]
  for (int l = 0; l < 5; ++l) {
    int V = Ss[l] * Ss[l] * Ss[l];
    dim3 grid((V + 31) / 32, (Cs[l] + 31) / 32);
    transpose_cv_kernel<<<grid, dim3(32, 8), 0, stream>>>(skip[l], vt[l], Cs[l], V);
  }

  // 2) repack weights into bf16 WMMA B-fragments
  {
    int t0 = (2 * HID / 16) * (FNUM / 32) * 512;      // 409600
    repack_w_kernel<<<(t0 + 255) / 256, 256, 0, stream>>>(w0, w0r, 2 * HID, FNUM, FNUM / 32, t0);
    int t1 = (HID / 16) * (2 * HID / 32) * 512;       // 131072
    repack_w_kernel<<<(t1 + 255) / 256, 256, 0, stream>>>(w1, w1r, HID, 2 * HID, (2 * HID) / 32, t1);
    int t2 = (HID / 16) * (HID / 32) * 512;           // 65536
    repack_w_kernel<<<(t2 + 255) / 256, 256, 0, stream>>>(w2, w2r, HID, HID, HID / 32, t2);
  }

  // 3) fused sample + MLP
  int blocks = (N + TILE_PTS - 1) / TILE_PTS;
  decoder_fused_kernel<<<blocks, 256, 0, stream>>>(
      vt[0], vt[1], vt[2], vt[3], vt[4], pts,
      w0r, w1r, w2r, b0, b1, b2, w3, b3, out, N);
}
